// RAJNIViTWrapper_47596827574882
// MI455X (gfx1250) — compile-verified, matrix operand order checked by reading, not run on hardware
//
#include <hip/hip_runtime.h>
#include <hip/hip_bf16.h>
#include <math.h>

typedef __attribute__((ext_vector_type(16))) __bf16 v16bf;
typedef __attribute__((ext_vector_type(8)))  __bf16 v8bf;
typedef __attribute__((ext_vector_type(4)))  __bf16 v4bf;
typedef __attribute__((ext_vector_type(8)))  float  v8f;
typedef __attribute__((ext_vector_type(4)))  unsigned int u32x4;
typedef __attribute__((ext_vector_type(8)))  int i32x8;
typedef __attribute__((ext_vector_type(4)))  int i32x4;

#if defined(__has_builtin)
#if __has_builtin(__builtin_amdgcn_tensor_load_to_lds)
#define HAVE_TDM 1
#endif
#endif

// ---------------- problem constants ----------------
constexpr int kB   = 64;
constexpr int kCh  = 3;
constexpr int kIMG = 224;
constexpr int kP   = 16;
constexpr int kD   = 768;
constexpr int kL   = 12;
constexpr int kNH  = 12;
constexpr int kFF  = 3072;
constexpr int kV   = 1000;
constexpr int kDH  = kD / kNH;     // 64
constexpr int kG   = kIMG / kP;    // 14
constexpr int kNP  = kG * kG;      // 196
constexpr int kN0  = kNP + 1;      // 197
constexpr float kEPS = 1e-6f;

constexpr int BM = 32, BN = 64, BK = 32;
constexpr int APITCH = 40;   // bf16 elements; 80B rows, 16B aligned, odd count of 16B units for banks
constexpr int BPITCH = 66;   // TDM-padded: 128B data + 4B pad per row

__device__ inline v4bf cvt4(float4 v) {
  v4bf r; r[0] = (__bf16)v.x; r[1] = (__bf16)v.y; r[2] = (__bf16)v.z; r[3] = (__bf16)v.w;
  return r;
}

// ============================================================
// Weight GEMM: C[M,N] = act(A[M,K]_f32 @ Bw[K,N]_bf16 + bias)
// A packed row-major pitch K (K % 32 == 0); Bw loaded by TDM.
// ============================================================
struct GemmW { const float* A; const __bf16* Bw; const float* bias; float* C;
               int M, N, K, act; };

__global__ __launch_bounds__(256) void gemm_tdm_bf16(GemmW p) {
  __shared__ __bf16 As[BM][APITCH];
  __shared__ __bf16 Bs[BK][BPITCH];

  const int tile_m = blockIdx.y * BM;
  const int tile_n = blockIdx.x * BN;
  const int tid  = threadIdx.x;
  const int lane = tid & 31;
  const int wave = tid >> 5;          // 8 waves: 2(M) x 4(N)
  const int wm = wave >> 2, wn = wave & 3;
  const int row16 = lane & 15;
  const int kb    = (lane >> 4) << 3; // 0 or 8

  v8f acc = {0.f,0.f,0.f,0.f,0.f,0.f,0.f,0.f};
  const bool fullA = (tile_m + BM <= p.M);
  const int  sr  = tid >> 3;          // 0..31
  const int  sc4 = (tid & 7) * 4;     // 0..28

  for (int k0 = 0; k0 < p.K; k0 += BK) {
#if HAVE_TDM
    if (tid < 32) {
      // --- Tensor DMA descriptor (D#): 2D bf16 tile BKxBN, LDS row pad 1DW/32DW ---
      unsigned lds_off = (unsigned)(size_t)&Bs[0][0];
      unsigned long long ga = (unsigned long long)(size_t)p.Bw +
                              ((unsigned long long)k0 * p.N + tile_n) * 2ull;
      u32x4 g0;
      g0[0] = 1u;                                            // count=1
      g0[1] = lds_off;                                       // lds_addr
      g0[2] = (unsigned)(ga & 0xffffffffu);                  // global_addr lo
      g0[3] = (unsigned)((ga >> 32) & 0x1ffffffu) | (2u << 30); // addr hi | type=2
      const int remN = p.N - tile_n;
      const int remK = p.K - k0;
      i32x8 g1;
      g1[0] = (1 << 16) | (1 << 20) | (4 << 22);             // 2B elems, pad_en, 32DW interval, 1DW pad
      g1[1] = (remN & 0xffff) << 16;                          // tensor_dim0 lo16
      g1[2] = ((remN >> 16) & 0xffff) | ((remK & 0xffff) << 16);
      g1[3] = ((remK >> 16) & 0xffff) | (BN << 16);           // tile_dim0=64
      g1[4] = BK;                                             // tile_dim1=32
      g1[5] = p.N;                                            // dim0 stride lo32
      g1[6] = 0; g1[7] = 0;
      i32x4 z4 = {0,0,0,0};
      i32x8 z8 = {0,0,0,0,0,0,0,0};
      __builtin_amdgcn_tensor_load_to_lds(g0, g1, z4, z4, z8, 0);
    }
#else
    for (int i = tid; i < BK * BN; i += 256) {
      int r = i >> 6, c = i & 63;
      int gn = tile_n + c;
      Bs[r][c] = (gn < p.N) ? p.Bw[(long)(k0 + r) * p.N + gn] : (__bf16)0.f;
    }
#endif
    // --- stage A (fp32 -> bf16), one float4 per thread ---
    if (fullA) {
      float4 v = *(const float4*)(p.A + (long)(tile_m + sr) * p.K + k0 + sc4);
      *(v4bf*)&As[sr][sc4] = cvt4(v);
    } else {
      const int gr = tile_m + sr;
      v4bf t4;
#pragma unroll
      for (int j = 0; j < 4; ++j) {
        float v = (gr < p.M) ? p.A[(long)gr * p.K + k0 + sc4 + j] : 0.f;
        t4[j] = (__bf16)v;
      }
      *(v4bf*)&As[sr][sc4] = t4;
    }
#if HAVE_TDM
    if (tid < 32) __builtin_amdgcn_s_wait_tensorcnt(0);
#endif
    __syncthreads();

    // --- fragments ---
    const int ar = wm * 16 + row16;
    v8bf a0 = *(const v8bf*)&As[ar][kb];
    v8bf a1 = *(const v8bf*)&As[ar][16 + kb];
    v16bf af = __builtin_shufflevector(a0, a1, 0,1,2,3,4,5,6,7,8,9,10,11,12,13,14,15);
    v16bf bfv;
    const int bc = wn * 16 + row16;
#pragma unroll
    for (int e = 0; e < 8; ++e) {
      bfv[e]     = Bs[kb + e][bc];
      bfv[e + 8] = Bs[16 + kb + e][bc];
    }
    acc = __builtin_amdgcn_wmma_f32_16x16x32_bf16(false, af, false, bfv,
                                                  (short)0, acc, false, false);
    __syncthreads();
  }

#pragma unroll
  for (int r = 0; r < 8; ++r) {
    int m = tile_m + wm * 16 + r + ((lane >> 4) << 3);
    int n = tile_n + wn * 16 + row16;
    if (m < p.M && n < p.N) {
      float v = acc[r];
      if (p.bias) v += p.bias[n];
      if (p.act == 1) {
        float x = v;
        v = 0.5f * x * (1.f + tanhf(0.7978845608028654f * (x + 0.044715f * x * x * x)));
      }
      p.C[(long)m * p.N + n] = v;
    }
  }
}

// ============================================================
// Attention GEMM (batched, strided, all-fp32 inputs -> bf16 tiles)
// C[z][m,n] = alpha * sum_k A[z][m,k] * B[z][k,n]
// B element: Bm[k*b_rs + n*b_cs]. Both LDS tiles stored K-major.
// ============================================================
struct GemmA { const float* A; const float* Bm; float* C;
               long a_rs, a_bz, a_hz;
               long b_rs, b_cs, b_bz, b_hz;
               long c_rs, c_bz, c_hz;
               int M, N, K, zdiv; float alpha; };

__global__ __launch_bounds__(256) void gemm_attn_bf16(GemmA p) {
  __shared__ __bf16 As[BM][APITCH];
  __shared__ __bf16 Bt[BN][APITCH];   // transposed: [n][k]

  const int z  = blockIdx.z;
  const int zb = z / p.zdiv, zh = z % p.zdiv;
  const float* A  = p.A  + (long)zb * p.a_bz + (long)zh * p.a_hz;
  const float* Bm = p.Bm + (long)zb * p.b_bz + (long)zh * p.b_hz;
  float*       C  = p.C  + (long)zb * p.c_bz + (long)zh * p.c_hz;

  const int tile_m = blockIdx.y * BM;
  const int tile_n = blockIdx.x * BN;
  const int tid  = threadIdx.x;
  const int lane = tid & 31;
  const int wave = tid >> 5;
  const int wm = wave >> 2, wn = wave & 3;
  const int row16 = lane & 15;
  const int kb    = (lane >> 4) << 3;

  v8f acc = {0.f,0.f,0.f,0.f,0.f,0.f,0.f,0.f};
  const bool aAligned = ((p.a_rs & 3) == 0);
  const bool bcs1 = (p.b_cs == 1);

  for (int k0 = 0; k0 < p.K; k0 += BK) {
    // ---- stage A ----
    {
      const int sr = tid >> 3, sc4 = (tid & 7) * 4;
      const int gr = tile_m + sr;
      if (aAligned && tile_m + BM <= p.M && k0 + BK <= p.K) {
        float4 v = *(const float4*)(A + (long)gr * p.a_rs + k0 + sc4);
        *(v4bf*)&As[sr][sc4] = cvt4(v);
      } else {
        v4bf t4;
#pragma unroll
        for (int j = 0; j < 4; ++j) {
          int gc = k0 + sc4 + j;
          float v = (gr < p.M && gc < p.K) ? A[(long)gr * p.a_rs + gc] : 0.f;
          t4[j] = (__bf16)v;
        }
        *(v4bf*)&As[sr][sc4] = t4;
      }
    }
    // ---- stage B (transposed into Bt[n][k]) ----
    const bool fullB = (tile_n + BN <= p.N) && (k0 + BK <= p.K);
    if (bcs1) {
      // rows contiguous along n: vector-load along n, scalar transpose stores
      const int rr = tid >> 4, cc4 = (tid & 15) * 4;
#pragma unroll
      for (int hh = 0; hh < 2; ++hh) {
        const int r = rr + 16 * hh;
        const int gk = k0 + r;
        if (fullB && ((p.b_rs & 3) == 0)) {
          float4 v = *(const float4*)(Bm + (long)gk * p.b_rs + tile_n + cc4);
          Bt[cc4 + 0][r] = (__bf16)v.x; Bt[cc4 + 1][r] = (__bf16)v.y;
          Bt[cc4 + 2][r] = (__bf16)v.z; Bt[cc4 + 3][r] = (__bf16)v.w;
        } else {
#pragma unroll
          for (int j = 0; j < 4; ++j) {
            int gn = tile_n + cc4 + j;
            float v = (gk < p.K && gn < p.N) ? Bm[(long)gk * p.b_rs + gn] : 0.f;
            Bt[cc4 + j][r] = (__bf16)v;
          }
        }
      }
    } else {
      // b_rs==1: contiguous along k: vector load+store along k
      const int c = tid & 63, r4 = (tid >> 6) * 4;
      const int gn = tile_n + c;
#pragma unroll
      for (int hh = 0; hh < 2; ++hh) {
        const int r = r4 + 16 * hh;
        if (fullB && ((p.b_cs & 3) == 0)) {
          float4 v = *(const float4*)(Bm + (long)gn * p.b_cs + k0 + r);
          *(v4bf*)&Bt[c][r] = cvt4(v);
        } else {
          v4bf t4;
#pragma unroll
          for (int j = 0; j < 4; ++j) {
            int gk = k0 + r + j;
            float v = (gn < p.N && gk < p.K) ? Bm[(long)gn * p.b_cs + gk] : 0.f;
            t4[j] = (__bf16)v;
          }
          *(v4bf*)&Bt[c][r] = t4;
        }
      }
    }
    __syncthreads();

    const int ar = wm * 16 + row16;
    const int bc = wn * 16 + row16;
    v8bf a0 = *(const v8bf*)&As[ar][kb];
    v8bf a1 = *(const v8bf*)&As[ar][16 + kb];
    v8bf b0 = *(const v8bf*)&Bt[bc][kb];
    v8bf b1 = *(const v8bf*)&Bt[bc][16 + kb];
    v16bf af = __builtin_shufflevector(a0, a1, 0,1,2,3,4,5,6,7,8,9,10,11,12,13,14,15);
    v16bf bf = __builtin_shufflevector(b0, b1, 0,1,2,3,4,5,6,7,8,9,10,11,12,13,14,15);
    acc = __builtin_amdgcn_wmma_f32_16x16x32_bf16(false, af, false, bf,
                                                  (short)0, acc, false, false);
    __syncthreads();
  }

#pragma unroll
  for (int r = 0; r < 8; ++r) {
    int m = tile_m + wm * 16 + r + ((lane >> 4) << 3);
    int n = tile_n + wn * 16 + row16;
    if (m < p.M && n < p.N)
      C[(long)m * p.c_rs + n] = acc[r] * p.alpha;
  }
}

// ---------------- fp32 -> bf16 convert ----------------
__global__ void cvt_bf16_kernel(const float* __restrict__ s, __bf16* __restrict__ d, long n) {
  long i = (long)blockIdx.x * blockDim.x + threadIdx.x;
  if (i < n) d[i] = (__bf16)s[i];
}

// ---------------- LayerNorm: one block per row ----------------
__global__ __launch_bounds__(256) void ln_kernel(const float* __restrict__ x, long xstride,
                                                 const float* __restrict__ g,
                                                 const float* __restrict__ b,
                                                 float* __restrict__ y, long ystride, int dim) {
  const long row = blockIdx.x;
  const float* xr = x + row * xstride;
  float* yr = y + row * ystride;
  __shared__ float red[256];
  const int t = threadIdx.x;

  float s = 0.f;
  for (int i = t; i < dim; i += 256) s += xr[i];
  red[t] = s; __syncthreads();
  for (int o = 128; o > 0; o >>= 1) { if (t < o) red[t] += red[t + o]; __syncthreads(); }
  const float mu = red[0] / dim;
  __syncthreads();

  s = 0.f;
  for (int i = t; i < dim; i += 256) { float d = xr[i] - mu; s += d * d; }
  red[t] = s; __syncthreads();
  for (int o = 128; o > 0; o >>= 1) { if (t < o) red[t] += red[t + o]; __syncthreads(); }
  const float rstd = rsqrtf(red[0] / dim + kEPS);

  for (int i = t; i < dim; i += 256)
    yr[i] = (xr[i] - mu) * rstd * g[i] + b[i];
}

// ---------------- row softmax ----------------
__global__ __launch_bounds__(256) void softmax_kernel(float* __restrict__ att, int n) {
  float* a = att + (long)blockIdx.x * n;
  __shared__ float red[256];
  const int t = threadIdx.x;

  float m = -3.0e38f;
  for (int i = t; i < n; i += 256) m = fmaxf(m, a[i]);
  red[t] = m; __syncthreads();
  for (int o = 128; o > 0; o >>= 1) { if (t < o) red[t] = fmaxf(red[t], red[t + o]); __syncthreads(); }
  const float mx = red[0];
  __syncthreads();

  float s = 0.f;
  for (int i = t; i < n; i += 256) { float e = __expf(a[i] - mx); a[i] = e; s += e; }
  red[t] = s; __syncthreads();
  for (int o = 128; o > 0; o >>= 1) { if (t < o) red[t] += red[t + o]; __syncthreads(); }
  const float inv = 1.f / red[0];
  __syncthreads();
  for (int i = t; i < n; i += 256) a[i] *= inv;
}

// ---------------- cls attention mean over heads ----------------
__global__ void cls_attn_kernel(const float* __restrict__ att, int nt, float* __restrict__ ca) {
  const int b = blockIdx.x;
  for (int j = threadIdx.x; j < nt - 1; j += blockDim.x) {
    float s = 0.f;
    for (int h = 0; h < kNH; ++h)
      s += att[((long)(b * kNH + h) * nt) * nt + (1 + j)];
    ca[(long)b * (nt - 1) + j] = s * (1.f / kNH);
  }
}

// ---------------- top-k (rank-based) + ascending compaction ----------------
__global__ __launch_bounds__(256) void topk_kernel(const float* __restrict__ ca, int n, int keep,
                                                   int* __restrict__ keep_idx) {
  const int b = blockIdx.x;
  const float* v = ca + (long)b * n;
  __shared__ unsigned char flag[256];
  for (int i = threadIdx.x; i < n; i += 256) {
    const float vi = v[i];
    int rank = 0;
    for (int j = 0; j < n; ++j) {
      const float vj = v[j];
      if (vj > vi || (vj == vi && j < i)) rank++;
    }
    flag[i] = (rank < keep) ? 1 : 0;
  }
  __syncthreads();
  if (threadIdx.x == 0) {
    int* out = keep_idx + (long)b * (keep + 1);
    out[0] = 0;
    int c = 1;
    for (int i = 0; i < n; ++i) if (flag[i]) out[c++] = i + 1;
  }
}

// ---------------- gather rows by keep_idx ----------------
__global__ void gather_kernel(const float* __restrict__ src, const int* __restrict__ idx,
                              float* __restrict__ dst, int newN, int oldN, long total) {
  long i = (long)blockIdx.x * blockDim.x + threadIdx.x;
  if (i >= total) return;
  int d = (int)(i % kD);
  int t = (int)((i / kD) % newN);
  int b = (int)(i / ((long)kD * newN));
  int sr = idx[(long)b * newN + t];
  dst[i] = src[((long)b * oldN + sr) * kD + d];
}

// ---------------- residual add ----------------
__global__ void add_kernel(float* __restrict__ h, const float* __restrict__ o, long total) {
  long i = (long)blockIdx.x * blockDim.x + threadIdx.x;
  if (i < total) h[i] += o[i];
}

// ---------------- patchify ----------------
__global__ void patchify_kernel(const float* __restrict__ x, float* __restrict__ p) {
  long i = (long)blockIdx.x * blockDim.x + threadIdx.x;
  const long total = (long)kB * kNP * (kCh * kP * kP);
  if (i >= total) return;
  int e  = (int)(i % (kCh * kP * kP));
  int pt = (int)((i / (kCh * kP * kP)) % kNP);
  int b  = (int)(i / ((long)(kCh * kP * kP) * kNP));
  int c  = e >> 8;
  int py = (e >> 4) & 15;
  int px = e & 15;
  int gy = pt / kG, gx = pt % kG;
  p[i] = x[((long)(b * kCh + c) * kIMG + (gy * kP + py)) * kIMG + (gx * kP + px)];
}

// ---------------- assemble h = [cls; patch_embed] + pos ----------------
__global__ void assemble_kernel(const float* __restrict__ hp, const float* __restrict__ cls,
                                const float* __restrict__ pos, float* __restrict__ h) {
  long i = (long)blockIdx.x * blockDim.x + threadIdx.x;
  const long total = (long)kB * kN0 * kD;
  if (i >= total) return;
  int d = (int)(i % kD);
  int t = (int)((i / kD) % kN0);
  int b = (int)(i / ((long)kD * kN0));
  float v = (t == 0) ? cls[d] : hp[((long)b * kNP + (t - 1)) * kD + d];
  h[i] = v + pos[(long)t * kD + d];
}

// ---------------- host helpers ----------------
static void launch_gemm_w(hipStream_t s, const float* A, const __bf16* Bw, const float* bias,
                          float* C, int M, int N, int K, int act) {
  GemmW p{A, Bw, bias, C, M, N, K, act};
  dim3 grid((N + BN - 1) / BN, (M + BM - 1) / BM, 1);
  gemm_tdm_bf16<<<grid, 256, 0, s>>>(p);
}

static void launch_gemm_a(hipStream_t s, const float* A, long a_rs, long a_bz, long a_hz,
                          const float* Bm, long b_rs, long b_cs, long b_bz, long b_hz,
                          float* C, long c_rs, long c_bz, long c_hz,
                          int M, int N, int K, int Z, int zdiv, float alpha) {
  GemmA p{A, Bm, C, a_rs, a_bz, a_hz, b_rs, b_cs, b_bz, b_hz, c_rs, c_bz, c_hz,
          M, N, K, zdiv, alpha};
  dim3 grid((N + BN - 1) / BN, (M + BM - 1) / BM, Z);
  gemm_attn_bf16<<<grid, 256, 0, s>>>(p);
}

static inline long up64(long n) { return (n + 63) & ~63L; }

extern "C" void kernel_launch(void* const* d_in, const int* in_sizes, int n_in,
                              void* d_out, int out_size, void* d_ws, size_t ws_size,
                              hipStream_t stream) {
  (void)in_sizes; (void)n_in; (void)out_size; (void)ws_size;

  const float* x         = (const float*)d_in[0];
  const float* patch_w   = (const float*)d_in[1];
  const float* patch_b   = (const float*)d_in[2];
  const float* cls_token = (const float*)d_in[3];
  const float* pos_embed = (const float*)d_in[4];
  const float* ln1_g     = (const float*)d_in[5];
  const float* ln1_b     = (const float*)d_in[6];
  const float* qkv_w     = (const float*)d_in[7];
  const float* qkv_b     = (const float*)d_in[8];
  const float* proj_w    = (const float*)d_in[9];
  const float* proj_b    = (const float*)d_in[10];
  const float* ln2_g     = (const float*)d_in[11];
  const float* ln2_b     = (const float*)d_in[12];
  const float* fc1_w     = (const float*)d_in[13];
  const float* fc1_b     = (const float*)d_in[14];
  const float* fc2_w     = (const float*)d_in[15];
  const float* fc2_b     = (const float*)d_in[16];
  const float* norm_g    = (const float*)d_in[17];
  const float* norm_b    = (const float*)d_in[18];
  const float* head_w    = (const float*)d_in[19];
  const float* head_b    = (const float*)d_in[20];
  float* out = (float*)d_out;

  // ---- carve workspace ----
  float* base = (float*)d_ws;
  long off = 0;
  auto carve    = [&](long n) { float* p = base + off; off += up64(n); return p; };
  auto carve_bf = [&](long n) { __bf16* p = (__bf16*)(base + off); off += up64((n + 1) / 2); return p; };

  const long szH   = (long)kB * kN0 * kD;
  const long szQKV = (long)kB * kN0 * 3 * kD;
  const long szATT = (long)kB * kNH * kN0 * kN0;
  const long szMLP = (long)kB * kN0 * kFF;

  float* hA    = carve(szH);
  float* hB    = carve(szH);
  float* xn    = carve(szH);
  float* qkv   = carve(szQKV);
  float* att   = carve(szATT);
  float* ao    = carve(szH);
  float* outA  = carve(szH);
  float* outB  = carve(szH);
  float* mlp1  = carve(szMLP);
  float* cattn = carve((long)kB * kNP);
  float* clsln = carve((long)kB * kD);
  int*   kidx  = (int*)carve((long)kB * kN0);
  // bf16 weight copies
  const long nPATCH = (long)(kCh * kP * kP) * kD;
  const long nQKV   = (long)kL * kD * 3 * kD;
  const long nPROJ  = (long)kL * kD * kD;
  const long nFC1   = (long)kL * kD * kFF;
  const long nFC2   = (long)kL * kFF * kD;
  const long nHEAD  = (long)kD * kV;
  __bf16* wPATCH = carve_bf(nPATCH);
  __bf16* wQKV   = carve_bf(nQKV);
  __bf16* wPROJ  = carve_bf(nPROJ);
  __bf16* wFC1   = carve_bf(nFC1);
  __bf16* wFC2   = carve_bf(nFC2);
  __bf16* wHEAD  = carve_bf(nHEAD);

  float* patches = mlp1;   // alias: used only before any MLP
  float* hp      = qkv;    // alias: used only before any QKV

  const int TPB = 256;
  auto blocks1d = [](long n) { return (unsigned)((n + 255) / 256); };

  // ---- weight conversion to bf16 ----
  cvt_bf16_kernel<<<blocks1d(nPATCH), TPB, 0, stream>>>(patch_w, wPATCH, nPATCH);
  cvt_bf16_kernel<<<blocks1d(nQKV),   TPB, 0, stream>>>(qkv_w,   wQKV,   nQKV);
  cvt_bf16_kernel<<<blocks1d(nPROJ),  TPB, 0, stream>>>(proj_w,  wPROJ,  nPROJ);
  cvt_bf16_kernel<<<blocks1d(nFC1),   TPB, 0, stream>>>(fc1_w,   wFC1,   nFC1);
  cvt_bf16_kernel<<<blocks1d(nFC2),   TPB, 0, stream>>>(fc2_w,   wFC2,   nFC2);
  cvt_bf16_kernel<<<blocks1d(nHEAD),  TPB, 0, stream>>>(head_w,  wHEAD,  nHEAD);

  // ---- patch embedding ----
  {
    long tot = (long)kB * kNP * (kCh * kP * kP);
    patchify_kernel<<<blocks1d(tot), TPB, 0, stream>>>(x, patches);
    launch_gemm_w(stream, patches, wPATCH, patch_b, hp, kB * kNP, kD, kCh * kP * kP, 0);
    long tot2 = (long)kB * kN0 * kD;
    assemble_kernel<<<blocks1d(tot2), TPB, 0, stream>>>(hp, cls_token, pos_embed, hA);
  }

  float* h  = hA; float* halt = hB;
  float* ob = outA; float* obalt = outB;
  int Nt = kN0;

  for (int i = 0; i < kL; ++i) {
    const long rows = (long)kB * Nt;

    ln_kernel<<<(unsigned)rows, TPB, 0, stream>>>(h, kD, ln1_g + (long)i * kD,
                                                  ln1_b + (long)i * kD, xn, kD, kD);
    // qkv = xn @ qkv_w[i] + qkv_b[i]
    launch_gemm_w(stream, xn, wQKV + (long)i * kD * 3 * kD, qkv_b + (long)i * 3 * kD,
                  qkv, (int)rows, 3 * kD, kD, 0);

    // att = (q @ k^T) / sqrt(DH)   (B read k-contiguous: b_rs=1, b_cs=3D)
    launch_gemm_a(stream,
                  qkv,      3 * kD, (long)Nt * 3 * kD, kDH,
                  qkv + kD, 1, 3 * kD, (long)Nt * 3 * kD, kDH,
                  att, Nt, (long)kNH * Nt * Nt, (long)Nt * Nt,
                  Nt, Nt, kDH, kB * kNH, kNH, 0.125f);

    softmax_kernel<<<(unsigned)((long)kB * kNH * Nt), TPB, 0, stream>>>(att, Nt);

    // ao = att @ v   (b_rs=3D, b_cs=1)
    launch_gemm_a(stream,
                  att, Nt, (long)kNH * Nt * Nt, (long)Nt * Nt,
                  qkv + 2 * kD, 3 * kD, 1, (long)Nt * 3 * kD, kDH,
                  ao, kD, (long)Nt * kD, kDH,
                  Nt, kDH, Nt, kB * kNH, kNH, 1.f);

    // ob = ao @ proj_w[i] + proj_b[i]
    launch_gemm_w(stream, ao, wPROJ + (long)i * kD * kD, proj_b + (long)i * kD,
                  ob, (int)rows, kD, kD, 0);

    if (i == 3 || i == 6 || i == 9) {
      cls_attn_kernel<<<kB, TPB, 0, stream>>>(att, Nt, cattn);
      const int n = Nt - 1;
      const int keep = (7 * n + 9) / 10;       // ceil(0.7*n)
      const int newN = keep + 1;
      topk_kernel<<<kB, TPB, 0, stream>>>(cattn, n, keep, kidx);
      long tot = (long)kB * newN * kD;
      gather_kernel<<<blocks1d(tot), TPB, 0, stream>>>(h,  kidx, halt,  newN, Nt, tot);
      gather_kernel<<<blocks1d(tot), TPB, 0, stream>>>(ob, kidx, obalt, newN, Nt, tot);
      float* t;
      t = h;  h  = halt;  halt  = t;
      t = ob; ob = obalt; obalt = t;
      Nt = newN;
    }

    const long rows2 = (long)kB * Nt;
    add_kernel<<<blocks1d(rows2 * kD), TPB, 0, stream>>>(h, ob, rows2 * kD);

    ln_kernel<<<(unsigned)rows2, TPB, 0, stream>>>(h, kD, ln2_g + (long)i * kD,
                                                   ln2_b + (long)i * kD, xn, kD, kD);
    launch_gemm_w(stream, xn, wFC1 + (long)i * kD * kFF, fc1_b + (long)i * kFF,
                  mlp1, (int)rows2, kFF, kD, 1);
    launch_gemm_w(stream, mlp1, wFC2 + (long)i * kFF * kD, fc2_b + (long)i * kD,
                  ob, (int)rows2, kD, kFF, 0);
    add_kernel<<<blocks1d(rows2 * kD), TPB, 0, stream>>>(h, ob, rows2 * kD);
  }

  ln_kernel<<<kB, TPB, 0, stream>>>(h, (long)Nt * kD, norm_g, norm_b, clsln, kD, kD);
  launch_gemm_w(stream, clsln, wHEAD, head_b, out, kB, kV, kD, 0);
}
